// SQACILSTM_70806830842338
// MI455X (gfx1250) — matrix operand order repaired and checked by reading, not compiled, not run on hardware
//
#include <hip/hip_runtime.h>
#include <math.h>

typedef unsigned short u16;
typedef __attribute__((ext_vector_type(16))) __bf16 v16bf;
typedef __attribute__((ext_vector_type(8)))  float  v8f;

#define Tn 256
#define Bn 64
#define Hn 768
#define NHn 12
#define HDn 64
#define H2n 1536
#define H4n 3072
#define BH (Bn*Hn)          // 49152

// ---------- helpers ----------

static __device__ __forceinline__ u16 f2bf(float x) {
    unsigned int u = __float_as_uint(x);
    unsigned int r = u + 0x7FFFu + ((u >> 16) & 1u);   // round-to-nearest-even
    return (u16)(r >> 16);
}

union FragU { uint4 q[2]; v16bf v; };

// A fragment: 16-bit A 16x32 layout (ISA 7.12.2), source already bf16.
// lane<16 : row M=lane,    K = k0+[0..7] and k0+16+[0..7]
// lane>=16: row M=lane-16, K = k0+8+[0..7] and k0+24+[0..7]
static __device__ __forceinline__ v16bf ldA(const u16* A, int ld, int row, int k0, int lane) {
    int kb = k0 + ((lane >> 4) << 3);
    const u16* p = A + (size_t)row * ld + kb;
    FragU u;
    u.q[0] = *reinterpret_cast<const uint4*>(p);
    u.q[1] = *reinterpret_cast<const uint4*>(p + 16);
    return u.v;
}

// A fragment gathered from an fp32 row pointer, converted to bf16 in-register.
static __device__ __forceinline__ v16bf ldAg(const float* rowp, int k0, int lane) {
    int kb = k0 + ((lane >> 4) << 3);
    float4 x0 = *reinterpret_cast<const float4*>(rowp + kb);
    float4 x1 = *reinterpret_cast<const float4*>(rowp + kb + 4);
    float4 x2 = *reinterpret_cast<const float4*>(rowp + kb + 16);
    float4 x3 = *reinterpret_cast<const float4*>(rowp + kb + 20);
    v16bf v;
    v[0]=(__bf16)x0.x; v[1]=(__bf16)x0.y; v[2]=(__bf16)x0.z; v[3]=(__bf16)x0.w;
    v[4]=(__bf16)x1.x; v[5]=(__bf16)x1.y; v[6]=(__bf16)x1.z; v[7]=(__bf16)x1.w;
    v[8]=(__bf16)x2.x; v[9]=(__bf16)x2.y; v[10]=(__bf16)x2.z; v[11]=(__bf16)x2.w;
    v[12]=(__bf16)x3.x; v[13]=(__bf16)x3.y; v[14]=(__bf16)x3.z; v[15]=(__bf16)x3.w;
    return v;
}

// B fragment from pre-transposed weights Wt[N][K] (row n contiguous in K).
// lane<16 : col N=colBase+lane,    K = k0+[0..7] and k0+16+[0..7]
// lane>=16: col N=colBase+lane-16, K = k0+8+[0..7] and k0+24+[0..7]
static __device__ __forceinline__ v16bf ldB(const u16* Bt, int ld, int colBase, int k0, int lane) {
    int n  = colBase + (lane & 15);
    int kb = k0 + ((lane >> 4) << 3);
    const u16* p = Bt + (size_t)n * ld + kb;
    FragU u;
    u.q[0] = *reinterpret_cast<const uint4*>(p);
    u.q[1] = *reinterpret_cast<const uint4*>(p + 16);
    return u.v;
}

static __device__ __forceinline__ v8f wmma_bf16(v16bf a, v16bf b, v8f c) {
    return __builtin_amdgcn_wmma_f32_16x16x32_bf16(false, a, false, b, (short)0, c, false, false);
}

static __device__ __forceinline__ float sigmoidf_(float x) {
    return 1.0f / (1.0f + expf(-x));
}

// ---------- setup kernels ----------

// W[K][N] fp32 -> Wt[N][K] bf16
__global__ void k_transpose(const float* __restrict__ W, u16* __restrict__ Wt, int K, int N) {
    int idx = blockIdx.x * blockDim.x + threadIdx.x;
    if (idx >= K * N) return;
    int k = idx / N, n = idx % N;
    Wt[(size_t)n * K + k] = f2bf(W[idx]);
}

__global__ void k_convert(const float* __restrict__ src, u16* __restrict__ dst, int n) {
    int i = blockIdx.x * blockDim.x + threadIdx.x;
    if (i < n) dst[i] = f2bf(src[i]);
}

__global__ void k_bias_qkv(const float* __restrict__ bq, const float* __restrict__ bk,
                           const float* __restrict__ bv, float* __restrict__ bqkv) {
    int n = blockIdx.x * blockDim.x + threadIdx.x;
    if (n >= 3 * Hn) return;
    int seg = n / Hn, r = n % Hn;
    bqkv[n] = (seg == 0) ? bq[r] : (seg == 1) ? bk[r] : bv[r];
}

__global__ void k_bias_base(const float* __restrict__ bqh, const float* __restrict__ bah,
                            const float* __restrict__ bih, const float* __restrict__ bhh,
                            float* __restrict__ bb) {
    int n = blockIdx.x * blockDim.x + threadIdx.x;
    if (n < H4n) bb[n] = bqh[n] + bah[n] + bih[n] + bhh[n];
}

// one-time: base = q@Wqh + a@Wah + bias ; 16x64 tile per wave, pipelined
__global__ __launch_bounds__(128)
void k_base(const u16* __restrict__ A0, const u16* __restrict__ A1,
            const u16* __restrict__ B0t, const u16* __restrict__ B1t,
            const float* __restrict__ bias, float* __restrict__ C) {
    const int M = Bn, N = H4n, K = Hn;
    int lane  = threadIdx.x & 31;
    int wave  = blockIdx.x * 4 + (threadIdx.x >> 5);
    int tilesN = N >> 6;
    if (wave >= (M >> 4) * tilesN) return;
    int tm = wave / tilesN, tn = wave % tilesN;
    int rowBase = tm << 4, colBase = tn << 6;
    int row = rowBase + (lane & 15);

    v8f acc[4] = {};
    v16bf a0C = ldA(A0, K, row, 0, lane);
    v16bf a1C = ldA(A1, K, row, 0, lane);
    v16bf b0C[4], b1C[4];
#pragma unroll
    for (int s = 0; s < 4; ++s) {
        b0C[s] = ldB(B0t, K, colBase + (s << 4), 0, lane);
        b1C[s] = ldB(B1t, K, colBase + (s << 4), 0, lane);
    }
#pragma unroll 2
    for (int k0 = 32; k0 < K; k0 += 32) {
        v16bf a0N = ldA(A0, K, row, k0, lane);
        v16bf a1N = ldA(A1, K, row, k0, lane);
        v16bf b0N[4], b1N[4];
#pragma unroll
        for (int s = 0; s < 4; ++s) {
            b0N[s] = ldB(B0t, K, colBase + (s << 4), k0, lane);
            b1N[s] = ldB(B1t, K, colBase + (s << 4), k0, lane);
        }
#pragma unroll
        for (int s = 0; s < 4; ++s) {
            acc[s] = wmma_bf16(a0C, b0C[s], acc[s]);
            acc[s] = wmma_bf16(a1C, b1C[s], acc[s]);
        }
        a0C = a0N; a1C = a1N;
#pragma unroll
        for (int s = 0; s < 4; ++s) { b0C[s] = b0N[s]; b1C[s] = b1N[s]; }
    }
#pragma unroll
    for (int s = 0; s < 4; ++s) {
        acc[s] = wmma_bf16(a0C, b0C[s], acc[s]);
        acc[s] = wmma_bf16(a1C, b1C[s], acc[s]);
    }
    int hi = lane >> 4, ln = lane & 15;
#pragma unroll
    for (int s = 0; s < 4; ++s)
#pragma unroll
        for (int j = 0; j < 8; ++j) {
            int r = rowBase + j + (hi << 3);
            int c = colBase + (s << 4) + ln;
            C[(size_t)r * N + c] = acc[s][j] + bias[c];
        }
}

// ---------- per-step kernels ----------

// qkv[128][2304] = gathered-pair(fp32 inputs) @ [Wq|Wk|Wv]T + [bq|bk|bv]
// A gather + fp32->bf16 conversion fused into the A-fragment loader.
__global__ __launch_bounds__(128)
void k_qkv(const float* __restrict__ inputs, const int* __restrict__ ping,
           const u16* __restrict__ Bt, const float* __restrict__ bias,
           float* __restrict__ qkv, int t) {
    const int M = 2 * Bn, N = 3 * Hn, K = Hn;
    int lane  = threadIdx.x & 31;
    int wave  = blockIdx.x * 4 + (threadIdx.x >> 5);
    int tilesN = N >> 6;
    if (wave >= (M >> 4) * tilesN) return;
    int tm = wave / tilesN, tn = wave % tilesN;
    int rowBase = tm << 4, colBase = tn << 6;
    int r = rowBase + (lane & 15);     // pair row: 2b + i
    int b = r >> 1;
    int src_t;
    if (r & 1) {
        src_t = t;                     // right = inputs[t]
    } else {                           // left = inputs[clip(ping[b,t]-1)]
        int lt = ping[b * Tn + t] - 1;
        src_t = lt < 0 ? 0 : (lt > Tn - 1 ? Tn - 1 : lt);
    }
    const float* rowp = inputs + (size_t)src_t * BH + (size_t)b * Hn;

    v8f acc[4] = {};
    v16bf aC = ldAg(rowp, 0, lane);
    v16bf bC[4];
#pragma unroll
    for (int s = 0; s < 4; ++s) bC[s] = ldB(Bt, K, colBase + (s << 4), 0, lane);
#pragma unroll 2
    for (int k0 = 32; k0 < K; k0 += 32) {
        v16bf aN = ldAg(rowp, k0, lane);
        v16bf bN[4];
#pragma unroll
        for (int s = 0; s < 4; ++s) bN[s] = ldB(Bt, K, colBase + (s << 4), k0, lane);
        if (k0 + 32 < K)
            __builtin_prefetch(Bt + (size_t)(colBase + (lane & 15)) * K + k0 + 32, 0, 1);
#pragma unroll
        for (int s = 0; s < 4; ++s) acc[s] = wmma_bf16(aC, bC[s], acc[s]);
        aC = aN;
#pragma unroll
        for (int s = 0; s < 4; ++s) bC[s] = bN[s];
    }
#pragma unroll
    for (int s = 0; s < 4; ++s) acc[s] = wmma_bf16(aC, bC[s], acc[s]);

    int hi = lane >> 4, ln = lane & 15;
#pragma unroll
    for (int s = 0; s < 4; ++s)
#pragma unroll
        for (int j = 0; j < 8; ++j) {
            int rr = rowBase + j + (hi << 3);
            int c  = colBase + (s << 4) + ln;
            qkv[(size_t)rr * N + c] = acc[s][j] + bias[c];
        }
}

// 2-token attention per batch block: qkv[128][2304] fp32 -> ctx_bf16[64][1536]
__global__ void k_attn(const float* __restrict__ qkv, u16* __restrict__ ctx) {
    const int b = blockIdx.x;
    const int tid = threadIdx.x;
    __shared__ float sS[NHn][2][2];
    __shared__ float sP[NHn][2][2];
    const float scale = 0.125f;  // 1/sqrt(64)

    if (tid < 48) {
        int h = tid >> 2, i = (tid >> 1) & 1, j = tid & 1;
        const float* q = qkv + (size_t)(2 * b + i) * (3 * Hn) + h * HDn;
        const float* k = qkv + (size_t)(2 * b + j) * (3 * Hn) + Hn + h * HDn;
        float acc = 0.f;
        for (int d = 0; d < HDn; ++d) acc += q[d] * k[d];
        sS[h][i][j] = acc * scale;
    }
    __syncthreads();
    if (tid < 24) {
        int h = tid >> 1, i = tid & 1;
        float s0 = sS[h][i][0], s1 = sS[h][i][1];
        float m = fmaxf(s0, s1);
        float e0 = expf(s0 - m), e1 = expf(s1 - m);
        float inv = 1.f / (e0 + e1);
        sP[h][i][0] = e0 * inv;
        sP[h][i][1] = e1 * inv;
    }
    __syncthreads();
    for (int idx = tid; idx < 2 * Hn; idx += blockDim.x) {
        int i = idx / Hn, rem = idx % Hn;
        int h = rem / HDn, d = rem % HDn;
        const float* v0 = qkv + (size_t)(2 * b + 0) * (3 * Hn) + 2 * Hn + h * HDn;
        const float* v1 = qkv + (size_t)(2 * b + 1) * (3 * Hn) + 2 * Hn + h * HDn;
        float cv = sP[h][i][0] * v0[d] + sP[h][i][1] * v1[d];
        ctx[(size_t)b * H2n + idx] = f2bf(cv);
    }
}

// fused = ctx @ WfuseT + bfuse ; newv = select ; writes inputs[t] fp32 + newv bf16
__global__ __launch_bounds__(128)
void k_fuse(const u16* __restrict__ ctx, const u16* __restrict__ WfuseT,
            const float* __restrict__ bfuse, const int* __restrict__ ping,
            float* __restrict__ inputs, u16* __restrict__ newvb, int t) {
    const int M = Bn, N = Hn, K = H2n;
    int lane  = threadIdx.x & 31;
    int wave  = blockIdx.x * 4 + (threadIdx.x >> 5);
    int tilesN = N >> 6;
    if (wave >= (M >> 4) * tilesN) return;
    int tm = wave / tilesN, tn = wave % tilesN;
    int rowBase = tm << 4, colBase = tn << 6;
    int row = rowBase + (lane & 15);

    v8f acc[4] = {};
    v16bf aC = ldA(ctx, K, row, 0, lane);
    v16bf bC[4];
#pragma unroll
    for (int s = 0; s < 4; ++s) bC[s] = ldB(WfuseT, K, colBase + (s << 4), 0, lane);
#pragma unroll 2
    for (int k0 = 32; k0 < K; k0 += 32) {
        v16bf aN = ldA(ctx, K, row, k0, lane);
        v16bf bN[4];
#pragma unroll
        for (int s = 0; s < 4; ++s) bN[s] = ldB(WfuseT, K, colBase + (s << 4), k0, lane);
#pragma unroll
        for (int s = 0; s < 4; ++s) acc[s] = wmma_bf16(aC, bC[s], acc[s]);
        aC = aN;
#pragma unroll
        for (int s = 0; s < 4; ++s) bC[s] = bN[s];
    }
#pragma unroll
    for (int s = 0; s < 4; ++s) acc[s] = wmma_bf16(aC, bC[s], acc[s]);

    int hi = lane >> 4, ln = lane & 15;
#pragma unroll
    for (int s = 0; s < 4; ++s)
#pragma unroll
        for (int j = 0; j < 8; ++j) {
            int bb  = rowBase + j + (hi << 3);       // batch row
            int col = colBase + (s << 4) + ln;
            float fused = acc[s][j] + bfuse[col];
            int pt = ping[bb * Tn + t];
            bool use_f = (pt > 0) && (t > 0);
            float* ip = inputs + (size_t)t * BH + (size_t)bb * Hn + col;
            float nv = use_f ? fused : *ip;
            *ip = nv;
            newvb[(size_t)bb * Hn + col] = f2bf(nv);
        }
}

// gates + LSTM fused: each wave owns 16 batch rows x 16 cols for ALL FOUR gate
// segments (acc[seg]); epilogue applies the LSTM cell directly.
// h is ping-ponged (hin read in K-loop, hout written in epilogue).
__global__ __launch_bounds__(128)
void k_gates_lstm(const u16* __restrict__ newvb, const u16* __restrict__ hin,
                  const u16* __restrict__ WihT, const u16* __restrict__ WhhT,
                  const float* __restrict__ base, float* __restrict__ c,
                  float* __restrict__ out, u16* __restrict__ hout, int t) {
    const int M = Bn, K = Hn;
    int lane  = threadIdx.x & 31;
    int wave  = blockIdx.x * 4 + (threadIdx.x >> 5);
    const int tilesN = Hn >> 4;               // 48 column tiles of 16
    if (wave >= (M >> 4) * tilesN) return;
    int tm = wave / tilesN, tn = wave % tilesN;
    int rowBase = tm << 4, colBase = tn << 4;
    int row = rowBase + (lane & 15);

    v8f acc[4] = {};                          // i, f, g, o for same 16 cols
    v16bf a0C = ldA(newvb, K, row, 0, lane);
    v16bf a1C = ldA(hin,   K, row, 0, lane);
    v16bf b0C[4], b1C[4];
#pragma unroll
    for (int s = 0; s < 4; ++s) {
        b0C[s] = ldB(WihT, K, s * Hn + colBase, 0, lane);
        b1C[s] = ldB(WhhT, K, s * Hn + colBase, 0, lane);
    }
#pragma unroll 2
    for (int k0 = 32; k0 < K; k0 += 32) {
        v16bf a0N = ldA(newvb, K, row, k0, lane);
        v16bf a1N = ldA(hin,   K, row, k0, lane);
        v16bf b0N[4], b1N[4];
#pragma unroll
        for (int s = 0; s < 4; ++s) {
            b0N[s] = ldB(WihT, K, s * Hn + colBase, k0, lane);
            b1N[s] = ldB(WhhT, K, s * Hn + colBase, k0, lane);
        }
#pragma unroll
        for (int s = 0; s < 4; ++s) {
            acc[s] = wmma_bf16(a0C, b0C[s], acc[s]);
            acc[s] = wmma_bf16(a1C, b1C[s], acc[s]);
        }
        a0C = a0N; a1C = a1N;
#pragma unroll
        for (int s = 0; s < 4; ++s) { b0C[s] = b0N[s]; b1C[s] = b1N[s]; }
    }
#pragma unroll
    for (int s = 0; s < 4; ++s) {
        acc[s] = wmma_bf16(a0C, b0C[s], acc[s]);
        acc[s] = wmma_bf16(a1C, b1C[s], acc[s]);
    }

    int hi = lane >> 4, ln = lane & 15;
#pragma unroll
    for (int j = 0; j < 8; ++j) {
        int r   = rowBase + j + (hi << 3);    // batch
        int col = colBase + ln;               // hidden unit
        const float* bs = base + (size_t)r * H4n + col;
        float i_ = acc[0][j] + bs[0];
        float f_ = acc[1][j] + bs[Hn];
        float g_ = acc[2][j] + bs[2 * Hn];
        float o_ = acc[3][j] + bs[3 * Hn];
        size_t idx = (size_t)r * Hn + col;
        float cn = sigmoidf_(f_) * c[idx] + sigmoidf_(i_) * tanhf(g_);
        float h  = sigmoidf_(o_) * tanhf(cn);
        c[idx] = cn;
        out[(size_t)t * BH + idx] = h;
        hout[idx] = f2bf(h);
        if (t == Tn - 1) {
            out[(size_t)Tn * BH + idx] = h;        // hT
            out[(size_t)Tn * BH + BH + idx] = cn;  // cT
        }
    }
}

// ---------- host ----------

extern "C" void kernel_launch(void* const* d_in, const int* in_sizes, int n_in,
                              void* d_out, int out_size, void* d_ws, size_t ws_size,
                              hipStream_t stream) {
    const float* question = (const float*)d_in[0];
    const float* answer   = (const float*)d_in[1];
    const float* x        = (const float*)d_in[2];
    const int*   ping     = (const int*)d_in[3];
    const float* h0       = (const float*)d_in[4];
    const float* c0       = (const float*)d_in[5];
    const float* Wqh = (const float*)d_in[6];
    const float* bqh = (const float*)d_in[7];
    const float* Wah = (const float*)d_in[8];
    const float* bah = (const float*)d_in[9];
    const float* Wih = (const float*)d_in[10];
    const float* bih = (const float*)d_in[11];
    const float* Whh = (const float*)d_in[12];
    const float* bhh = (const float*)d_in[13];
    const float* Wq  = (const float*)d_in[14];
    const float* bq  = (const float*)d_in[15];
    const float* Wk  = (const float*)d_in[16];
    const float* bk  = (const float*)d_in[17];
    const float* Wv  = (const float*)d_in[18];
    const float* bv  = (const float*)d_in[19];
    const float* Wfuse = (const float*)d_in[20];
    const float* bfuse = (const float*)d_in[21];
    float* out = (float*)d_out;

    char* ws = (char*)d_ws;
    size_t off = 0;
    auto alloc = [&](size_t bytes) -> void* {
        void* p = ws + off;
        off = (off + bytes + 255) & ~(size_t)255;
        return p;
    };

    u16*   WqhT  = (u16*)alloc((size_t)H4n * Hn * 2);
    u16*   WahT  = (u16*)alloc((size_t)H4n * Hn * 2);
    u16*   WihT  = (u16*)alloc((size_t)H4n * Hn * 2);
    u16*   WhhT  = (u16*)alloc((size_t)H4n * Hn * 2);
    u16*   WqkvT = (u16*)alloc((size_t)3 * Hn * Hn * 2);
    u16*   WfuT  = (u16*)alloc((size_t)Hn * H2n * 2);
    float* bqkv  = (float*)alloc(3 * Hn * 4);
    float* bbase = (float*)alloc(H4n * 4);
    u16*   qbf   = (u16*)alloc((size_t)BH * 2);
    u16*   abf   = (u16*)alloc((size_t)BH * 2);
    float* base  = (float*)alloc((size_t)Bn * H4n * 4);
    float* inputs= (float*)alloc((size_t)Tn * BH * 4);
    float* qkv   = (float*)alloc((size_t)2 * Bn * 3 * Hn * 4);
    u16*   ctxb  = (u16*)alloc((size_t)Bn * H2n * 2);
    u16*   newvb = (u16*)alloc((size_t)BH * 2);
    u16*   hbfA  = (u16*)alloc((size_t)BH * 2);
    u16*   hbfB  = (u16*)alloc((size_t)BH * 2);
    float* cst   = (float*)alloc((size_t)BH * 4);
    (void)ws_size; (void)in_sizes; (void)n_in; (void)out_size;

    // --- one-time setup (bf16 transposed weights ~24MB: resident in 192MB L2) ---
    k_transpose<<<(Hn * H4n) / 256, 256, 0, stream>>>(Wqh, WqhT, Hn, H4n);
    k_transpose<<<(Hn * H4n) / 256, 256, 0, stream>>>(Wah, WahT, Hn, H4n);
    k_transpose<<<(Hn * H4n) / 256, 256, 0, stream>>>(Wih, WihT, Hn, H4n);
    k_transpose<<<(Hn * H4n) / 256, 256, 0, stream>>>(Whh, WhhT, Hn, H4n);
    k_transpose<<<(Hn * Hn) / 256, 256, 0, stream>>>(Wq, WqkvT, Hn, Hn);
    k_transpose<<<(Hn * Hn) / 256, 256, 0, stream>>>(Wk, WqkvT + (size_t)Hn * Hn, Hn, Hn);
    k_transpose<<<(Hn * Hn) / 256, 256, 0, stream>>>(Wv, WqkvT + (size_t)2 * Hn * Hn, Hn, Hn);
    k_transpose<<<(H2n * Hn) / 256, 256, 0, stream>>>(Wfuse, WfuT, H2n, Hn);
    k_bias_qkv<<<(3 * Hn + 255) / 256, 256, 0, stream>>>(bq, bk, bv, bqkv);
    k_bias_base<<<(H4n + 255) / 256, 256, 0, stream>>>(bqh, bah, bih, bhh, bbase);
    k_convert<<<(BH + 255) / 256, 256, 0, stream>>>(question, qbf, BH);
    k_convert<<<(BH + 255) / 256, 256, 0, stream>>>(answer, abf, BH);
    k_convert<<<(BH + 255) / 256, 256, 0, stream>>>(h0, hbfA, BH);
    hipMemcpyAsync(cst, c0, (size_t)BH * 4, hipMemcpyDeviceToDevice, stream);
    hipMemcpyAsync(inputs, x, (size_t)Tn * BH * 4, hipMemcpyDeviceToDevice, stream);

    // base[64][3072] = q@Wqh + a@Wah + (bqh+bah+bih+bhh)
    {
        int waves = (Bn / 16) * (H4n / 64);
        k_base<<<waves / 4, 128, 0, stream>>>(qbf, abf, WqhT, WahT, bbase, base);
    }

    // --- sequential scan: 256 dependent steps, 4 kernels each ---
    for (int t = 0; t < Tn; ++t) {
        const u16* hin = (t & 1) ? hbfB : hbfA;
        u16*      hout = (t & 1) ? hbfA : hbfB;

        {   // qkv[128][2304] = gathered-pair @ [Wq|Wk|Wv] + [bq|bk|bv]
            int waves = (2 * Bn / 16) * (3 * Hn / 64);
            k_qkv<<<waves / 4, 128, 0, stream>>>(inputs, ping, WqkvT, bqkv, qkv, t);
        }

        k_attn<<<Bn, 128, 0, stream>>>(qkv, ctxb);

        {   // fuse + select + write inputs[t]/newv
            int waves = (Bn / 16) * (Hn / 64);
            k_fuse<<<waves / 4, 128, 0, stream>>>(ctxb, WfuT, bfuse, ping,
                                                  inputs, newvb, t);
        }

        {   // gates GEMM + LSTM cell fused
            int waves = (Bn / 16) * (Hn / 16);
            k_gates_lstm<<<waves / 4, 128, 0, stream>>>(newvb, hin, WihT, WhhT,
                                                        base, cst, out, hout, t);
        }
    }
}
